// PanRepRGCN_11536282157488
// MI455X (gfx1250) — matrix operand order repaired and checked by reading, not compiled, not run on hardware
//
#include <hip/hip_runtime.h>

#define DD 64   // feature dim

typedef __attribute__((ext_vector_type(16))) _Float16 v16h;
typedef __attribute__((ext_vector_type(8)))  _Float16 v8h;
typedef __attribute__((ext_vector_type(8)))  float    v8f;

// ---------------------------------------------------------------------------
// Utility: zero a float buffer (deterministic init of accumulators each call)
// ---------------------------------------------------------------------------
__global__ void fill0_kernel(float* __restrict__ p, size_t n) {
    size_t t = (size_t)blockIdx.x * blockDim.x + threadIdx.x;
    if (t < n) p[t] = 0.0f;
}

// ---------------------------------------------------------------------------
// Convert f32 -> f16 (optionally ReLU first). Feeds the WMMA A-operand buffer.
// ---------------------------------------------------------------------------
__global__ void cvt_f16_kernel(const float* __restrict__ in,
                               _Float16* __restrict__ out, int n, int relu) {
    int t = blockIdx.x * blockDim.x + threadIdx.x;
    if (t >= n) return;
    float v = in[t];
    if (relu && v < 0.0f) v = 0.0f;
    out[t] = (_Float16)v;
}

// ---------------------------------------------------------------------------
// Combine bases with wcomp (W_r = sum_b wcomp[r,b]*bases[b]) and pack into the
// per-lane WMMA B-operand layout for v_wmma_f32_16x16x32_f16:
//   lane L owns column n = (L&15); K block: kk*32 + (e>>3)*16 + (L>>4)*8 + (e&7), e = 0..15
// Packed as wpk[((r*4+nt)*2+kk)*32 + lane][16 halves] -> 32 contiguous bytes
// per lane.  wcomp == nullptr packs a plain DxD matrix (decoder W1).
// ---------------------------------------------------------------------------
__global__ void pack_weights_kernel(const float* __restrict__ wsrc,
                                    const float* __restrict__ wcomp,
                                    int nrel, int nbases,
                                    _Float16* __restrict__ wpk) {
    int t = blockIdx.x * blockDim.x + threadIdx.x;
    int total = nrel * DD * DD;
    if (t >= total) return;
    int n = t & (DD - 1);
    int k = (t >> 6) & (DD - 1);
    int r = t >> 12;
    float acc = 0.0f;
    if (wcomp) {
        for (int b = 0; b < nbases; ++b)
            acc += wcomp[r * nbases + b] * wsrc[((size_t)b * DD + k) * DD + n];
    } else {
        acc = wsrc[(size_t)k * DD + n];
    }
    int nt   = n >> 4;
    int kk   = k >> 5;
    int lane = (n & 15) | (((k >> 3) & 1) << 4);
    int e    = (k & 7) | (((k >> 4) & 1) << 3);
    wpk[((((r * 4 + nt) * 2 + kk) * 32) + lane) * 16 + e] = (_Float16)acc;
}

// ---------------------------------------------------------------------------
// Batched GEMM: Ht[r, m, :] = h16[m, :] @ W_r  (f16 output -> L2-resident Ht).
// One wave per 16-row node tile; A tile loaded once, reused for all NREL x 4
// output subtiles.  NREL templated so the relation loop fully unrolls and the
// 8 B-operand loads per relation clause ahead of 8 back-to-back WMMAs.
// A layout per ISA: lane L holds row (L&15); elems 0..7 = K{half*8+0..7},
// elems 8..15 = K{16+half*8+0..7}, half = L>>4.
// ---------------------------------------------------------------------------
template <int NREL>
__global__ __launch_bounds__(32)
void rgcn_gemm_kernel(const _Float16* __restrict__ h16,
                      const _Float16* __restrict__ wpk,
                      _Float16* __restrict__ Ht, int nnodes) {
    int tile = blockIdx.x;
    int lane = threadIdx.x;
    int half = lane >> 4;
    int m    = (tile << 4) + (lane & 15);
    int mc   = m < nnodes ? m : nnodes - 1;
    bool fullTile = ((tile << 4) + 16) <= nnodes;   // wave-uniform

    const _Float16* arow = h16 + (size_t)mc * DD;
    v8h a0lo = *(const v8h*)(arow +      half * 8);
    v8h a0hi = *(const v8h*)(arow + 16 + half * 8);
    v8h a1lo = *(const v8h*)(arow + 32 + half * 8);
    v8h a1hi = *(const v8h*)(arow + 48 + half * 8);
    v16h a0 = __builtin_shufflevector(a0lo, a0hi, 0,1,2,3,4,5,6,7,8,9,10,11,12,13,14,15);
    v16h a1 = __builtin_shufflevector(a1lo, a1hi, 0,1,2,3,4,5,6,7,8,9,10,11,12,13,14,15);

#pragma unroll
    for (int r = 0; r < NREL; ++r) {
        // clause-load all 8 B operands for this relation (4 n-tiles x 2 K-steps)
        v16h b[8];
#pragma unroll
        for (int nt = 0; nt < 4; ++nt) {
            const _Float16* bp = wpk + ((((r * 4 + nt) * 2) * 32) + lane) * 16;
            b[2 * nt]     = *(const v16h*)bp;
            b[2 * nt + 1] = *(const v16h*)(bp + 512);
        }
        // 8 WMMAs: 4 independent accumulators, 2 chained K-steps each
        v8f c[4];
#pragma unroll
        for (int nt = 0; nt < 4; ++nt) {
            v8f z = {};
            z = __builtin_amdgcn_wmma_f32_16x16x32_f16(false, a0, false, b[2 * nt],
                                                       (short)0, z, false, false);
            c[nt] = __builtin_amdgcn_wmma_f32_16x16x32_f16(false, a1, false, b[2 * nt + 1],
                                                           (short)0, z, false, false);
        }
        _Float16* out = Ht + ((size_t)r * nnodes + ((size_t)tile << 4)) * DD;
        int colb = lane & 15;
        if (fullTile) {
#pragma unroll
            for (int nt = 0; nt < 4; ++nt)
#pragma unroll
                for (int v = 0; v < 8; ++v)
                    out[(size_t)(v + half * 8) * DD + nt * 16 + colb] = (_Float16)c[nt][v];
        } else {
#pragma unroll
            for (int nt = 0; nt < 4; ++nt)
#pragma unroll
                for (int v = 0; v < 8; ++v) {
                    int grow = (tile << 4) + v + half * 8;
                    if (grow < nnodes)
                        out[(size_t)(v + half * 8) * DD + nt * 16 + colb] = (_Float16)c[nt][v];
                }
        }
    }
}

// ---------------------------------------------------------------------------
// Decoder GEMM with fused bias + ReLU: d1 = relu(henc16 @ W1 + b1)  (f32 out)
// ---------------------------------------------------------------------------
__global__ __launch_bounds__(32)
void dec_gemm_kernel(const _Float16* __restrict__ h16,
                     const _Float16* __restrict__ wpk,
                     const float* __restrict__ bias,
                     float* __restrict__ d1, int nnodes) {
    int tile = blockIdx.x;
    int lane = threadIdx.x;
    int half = lane >> 4;
    int m    = (tile << 4) + (lane & 15);
    int mc   = m < nnodes ? m : nnodes - 1;
    bool fullTile = ((tile << 4) + 16) <= nnodes;

    const _Float16* arow = h16 + (size_t)mc * DD;
    v8h a0lo = *(const v8h*)(arow +      half * 8);
    v8h a0hi = *(const v8h*)(arow + 16 + half * 8);
    v8h a1lo = *(const v8h*)(arow + 32 + half * 8);
    v8h a1hi = *(const v8h*)(arow + 48 + half * 8);
    v16h a0 = __builtin_shufflevector(a0lo, a0hi, 0,1,2,3,4,5,6,7,8,9,10,11,12,13,14,15);
    v16h a1 = __builtin_shufflevector(a1lo, a1hi, 0,1,2,3,4,5,6,7,8,9,10,11,12,13,14,15);

    v16h b[8];
#pragma unroll
    for (int nt = 0; nt < 4; ++nt) {
        const _Float16* bp = wpk + (((nt * 2) * 32) + lane) * 16;
        b[2 * nt]     = *(const v16h*)bp;
        b[2 * nt + 1] = *(const v16h*)(bp + 512);
    }
    v8f c[4];
#pragma unroll
    for (int nt = 0; nt < 4; ++nt) {
        v8f z = {};
        z = __builtin_amdgcn_wmma_f32_16x16x32_f16(false, a0, false, b[2 * nt],
                                                   (short)0, z, false, false);
        c[nt] = __builtin_amdgcn_wmma_f32_16x16x32_f16(false, a1, false, b[2 * nt + 1],
                                                       (short)0, z, false, false);
    }
    float* out = d1 + ((size_t)tile << 4) * DD;
    int colb = lane & 15;
#pragma unroll
    for (int nt = 0; nt < 4; ++nt) {
        float bv = bias[nt * 16 + colb];
        if (fullTile) {
#pragma unroll
            for (int v = 0; v < 8; ++v) {
                float val = c[nt][v] + bv;
                out[(size_t)(v + half * 8) * DD + nt * 16 + colb] = val > 0.0f ? val : 0.0f;
            }
        } else {
#pragma unroll
            for (int v = 0; v < 8; ++v) {
                int grow = (tile << 4) + v + half * 8;
                if (grow < nnodes) {
                    float val = c[nt][v] + bv;
                    out[(size_t)(v + half * 8) * DD + nt * 16 + colb] = val > 0.0f ? val : 0.0f;
                }
            }
        }
    }
}

// ---------------------------------------------------------------------------
// Edge gather/scale/scatter-add: the bandwidth-bound phase.
// Ht is f16 (102 MB, fits the 192 MB L2).  One wave per edge; each lane reads
// 2 f16 columns (4B coalesced), scales by norm in f32, two f32 atomics into
// the L2-resident accumulator.
// ---------------------------------------------------------------------------
__global__ void edge_scatter_kernel(const _Float16* __restrict__ Ht,
                                    const int* __restrict__ src,
                                    const int* __restrict__ dst,
                                    const int* __restrict__ etype,
                                    const float* __restrict__ norm,
                                    float* __restrict__ out,
                                    int nedges, int nnodes) {
    int t = blockIdx.x * blockDim.x + threadIdx.x;
    int e = t >> 5;
    int lane = t & 31;
    if (e >= nedges) return;
    int   s  = src[e];
    int   d  = dst[e];
    int   r  = etype[e];
    float nr = norm[e];
    const _Float16* mrow = Ht + ((size_t)r * nnodes + s) * DD + lane * 2;
    float m0 = (float)mrow[0];
    float m1 = (float)mrow[1];
    float* orow = out + (size_t)d * DD + lane * 2;
    atomicAdd(orow,     m0 * nr);
    atomicAdd(orow + 1, m1 * nr);
}

// ---------------------------------------------------------------------------
// Final decoder: rec[n] = dot(d1[n,:], w2) + b2, wave-per-node reduction.
// ---------------------------------------------------------------------------
__global__ void dec_dot_kernel(const float* __restrict__ d1,
                               const float* __restrict__ w2,
                               const float* __restrict__ b2,
                               float* __restrict__ rec, int nnodes) {
    int gt = blockIdx.x * blockDim.x + threadIdx.x;
    int n = gt >> 5;
    int lane = gt & 31;
    if (n >= nnodes) return;
    const float* row = d1 + (size_t)n * DD;
    float s = row[lane] * w2[lane] + row[lane + 32] * w2[lane + 32];
#pragma unroll
    for (int off = 16; off > 0; off >>= 1)
        s += __shfl_xor(s, off, 32);
    if (lane == 0) rec[n] = s + b2[0];
}

// ---------------------------------------------------------------------------
// Host launch
// ---------------------------------------------------------------------------
extern "C" void kernel_launch(void* const* d_in, const int* in_sizes, int n_in,
                              void* d_out, int out_size, void* d_ws, size_t ws_size,
                              hipStream_t stream) {
    const float* h      = (const float*)d_in[0];
    const int*   src    = (const int*)d_in[1];
    const int*   dst    = (const int*)d_in[2];
    const int*   etype  = (const int*)d_in[3];
    const float* norm   = (const float*)d_in[4];
    const float* bases[3] = {(const float*)d_in[5], (const float*)d_in[7], (const float*)d_in[9]};
    const float* wcomp[3] = {(const float*)d_in[6], (const float*)d_in[8], (const float*)d_in[10]};
    const float* dec_w1 = (const float*)d_in[11];
    const float* dec_b1 = (const float*)d_in[12];
    const float* dec_w2 = (const float*)d_in[13];
    const float* dec_b2 = (const float*)d_in[14];

    const int N = in_sizes[0] / DD;   // 100000
    const int E = in_sizes[1];        // 1600000
    const int R = 8, NB = 4;

    // workspace layout (Ht now f16: R*N*DD*2 = 102.4 MB, L2-resident)
    char*  ws     = (char*)d_ws;
    size_t offH16 = 0;
    size_t offHt  = ((size_t)N * DD * sizeof(_Float16) + 255) & ~(size_t)255;
    size_t offBuf = offHt + ((((size_t)R * N * DD * sizeof(_Float16)) + 255) & ~(size_t)255);
    size_t offWpk = offBuf + ((((size_t)N * DD * sizeof(float)) + 255) & ~(size_t)255);
    _Float16* h16 = (_Float16*)(ws + offH16);
    _Float16* Ht  = (_Float16*)(ws + offHt);
    float*    buf = (float*)(ws + offBuf);   // layer accumulator / decoder d1
    _Float16* wpk = (_Float16*)(ws + offWpk);

    float* outF = (float*)d_out;   // [0..N): rec, [N..N+N*DD): henc
    float* henc = outF + N;

    const int tiles = (N + 15) / 16;
    const int ND    = N * DD;

    // zero rec + henc accumulators (deterministic per call)
    {
        size_t n = (size_t)out_size;
        fill0_kernel<<<(unsigned)((n + 255) / 256), 256, 0, stream>>>(outF, n);
    }

    // h -> f16 for WMMA A operand
    cvt_f16_kernel<<<(ND + 255) / 256, 256, 0, stream>>>(h, h16, ND, 0);

    for (int L = 0; L < 3; ++L) {
        pack_weights_kernel<<<(R * DD * DD + 255) / 256, 256, 0, stream>>>(
            bases[L], wcomp[L], R, NB, wpk);
        rgcn_gemm_kernel<8><<<tiles, 32, 0, stream>>>(h16, wpk, Ht, N);

        float* acc = (L == 2) ? henc : buf;
        if (L != 2) {
            size_t n = (size_t)ND;
            fill0_kernel<<<(unsigned)((n + 255) / 256), 256, 0, stream>>>(buf, n);
        }
        size_t ethreads = (size_t)E * 32;
        edge_scatter_kernel<<<(unsigned)((ethreads + 255) / 256), 256, 0, stream>>>(
            Ht, src, dst, etype, norm, acc, E, N);

        // prepare next-layer f16 input: ReLU for layers 0/1, plain for henc
        cvt_f16_kernel<<<(ND + 255) / 256, 256, 0, stream>>>(
            acc, h16, ND, (L == 2) ? 0 : 1);
    }

    // decoder: d1 = relu(henc @ W1 + b1); rec = d1 @ w2 + b2
    pack_weights_kernel<<<(DD * DD + 255) / 256, 256, 0, stream>>>(
        dec_w1, nullptr, 1, 1, wpk);
    dec_gemm_kernel<<<tiles, 32, 0, stream>>>(h16, wpk, dec_b1, buf, N);
    dec_dot_kernel<<<(N * 32 + 255) / 256, 256, 0, stream>>>(
        buf, dec_w2, dec_b2, outF, N);
}
